// FusedBaichuanLowBitMultiDecoderlayer_68221260529904
// MI455X (gfx1250) — matrix operand order repaired and checked by reading, not compiled, not run on hardware
//
#include <hip/hip_runtime.h>

#define SEQ    1024
#define HIDDEN 4096
#define NHEADS 32
#define HDIM   128
#define INTER  11008
#define THREEH 12288
#define NLAYERS 2

#if defined(__AMDGCN__) && __has_builtin(__builtin_amdgcn_global_load_async_to_lds_b128)
#define HAS_ASYNC 1
#endif

typedef __attribute__((ext_vector_type(16))) __bf16 v16bf;
typedef __attribute__((ext_vector_type(8)))  __bf16 v8bf;
typedef __attribute__((ext_vector_type(8)))  float  v8f;

static __device__ __forceinline__ __bf16 f2bf(float f) {
  union { float f; unsigned u; } a; a.f = f;
  unsigned r = a.u + 0x7FFFu + ((a.u >> 16) & 1u);   // round-to-nearest-even
  union { unsigned short s; __bf16 b; } o; o.s = (unsigned short)(r >> 16);
  return o.b;
}

// Direct global -> LDS DMA (ASYNCcnt) when available; VGPR round-trip otherwise.
// Builtin signature (from compiler diagnostic): (v4i addrspace(1)*, v4i addrspace(3)*, Ii, Ii)
static __device__ __forceinline__ void async_b128(const void* g, void* l) {
#ifdef HAS_ASYNC
  typedef __attribute__((__vector_size__(16))) int v4i;
  typedef __attribute__((address_space(1))) v4i gv4i_t;
  typedef __attribute__((address_space(3))) v4i lv4i_t;
  __builtin_amdgcn_global_load_async_to_lds_b128((gv4i_t*)g, (lv4i_t*)l, 0, 0);
#else
  *(uint4*)l = *(const uint4*)g;
#endif
}

static __device__ __forceinline__ void wait_async() {
#ifdef HAS_ASYNC
#if __has_builtin(__builtin_amdgcn_s_wait_asynccnt)
  __builtin_amdgcn_s_wait_asynccnt(0);
#else
  asm volatile("s_wait_asynccnt 0x0" ::: "memory");
#endif
#endif
}

// Assemble a 32-byte WMMA operand from two 16-byte chunks (off2 in halves).
static __device__ __forceinline__ v16bf ld_frag(const __bf16* p, int off2) {
  union { v16bf v; uint4 u[2]; } t;
  t.u[0] = *(const uint4*)(p);
  t.u[1] = *(const uint4*)(p + off2);
  return t.v;
}

static __device__ __forceinline__ v8bf cvt8(float4 a, float4 b) {
  v8bf r;
  r[0] = f2bf(a.x); r[1] = f2bf(a.y); r[2] = f2bf(a.z); r[3] = f2bf(a.w);
  r[4] = f2bf(b.x); r[5] = f2bf(b.y); r[6] = f2bf(b.z); r[7] = f2bf(b.w);
  return r;
}

static __device__ __forceinline__ v8f wmma_bf16(v16bf a, v16bf b, v8f c) {
  return __builtin_amdgcn_wmma_f32_16x16x32_bf16(false, a, false, b, (short)0, c, false, false);
}

static __device__ __forceinline__ v8f vzero8() {
  v8f z = {0.f, 0.f, 0.f, 0.f, 0.f, 0.f, 0.f, 0.f};
  return z;
}

// ---------------------------------------------------------------------------
__global__ void copy_f32(const float* __restrict__ src, float* __restrict__ dst) {
  size_t i = ((size_t)blockIdx.x * 256 + threadIdx.x) * 4;
  *(float4*)(dst + i) = *(const float4*)(src + i);
}

// ---------------------------------------------------------------------------
// RMSNorm: one block per token row, output bf16 activations.
__global__ __launch_bounds__(256) void rmsnorm_bf16(const float* __restrict__ X,
                                                    const float* __restrict__ w,
                                                    __bf16* __restrict__ Y) {
  __shared__ float red[256];
  const int t = threadIdx.x;
  const int row = blockIdx.x;
  const float* x = X + (size_t)row * HIDDEN;
  float4 v[4];
  float ss = 0.f;
#pragma unroll
  for (int i = 0; i < 4; ++i) {
    v[i] = *(const float4*)(x + i * 1024 + t * 4);
    ss += v[i].x * v[i].x + v[i].y * v[i].y + v[i].z * v[i].z + v[i].w * v[i].w;
  }
  red[t] = ss;
  __syncthreads();
  for (int s = 128; s > 0; s >>= 1) {
    if (t < s) red[t] += red[t + s];
    __syncthreads();
  }
  const float inv = rsqrtf(red[0] * (1.f / HIDDEN) + 1e-6f);
  __bf16* y = Y + (size_t)row * HIDDEN;
#pragma unroll
  for (int i = 0; i < 4; ++i) {
    const int c = i * 1024 + t * 4;
    float4 wv = *(const float4*)(w + c);
    y[c + 0] = f2bf(v[i].x * inv * wv.x);
    y[c + 1] = f2bf(v[i].y * inv * wv.y);
    y[c + 2] = f2bf(v[i].z * inv * wv.z);
    y[c + 3] = f2bf(v[i].w * inv * wv.w);
  }
}

// ---------------------------------------------------------------------------
// C[M,N] = A[M,K](bf16) * W[N,K]^T(fp32->bf16 in regs), double-buffered LDS,
// 128x128 block tile, 8 waves, 64x32 per wave, BK=32, row pad -> 40 halves.
// A tile staged with global_load_async_to_lds_b128 (ASYNCcnt).
__global__ __launch_bounds__(256) void gemm_bf16(const __bf16* __restrict__ A,
                                                 const float* __restrict__ W,
                                                 float* C, const float* resid,
                                                 int N, int K, int add_resid) {
  __shared__ __bf16 As[2][128 * 40];
  __shared__ __bf16 Bs[2][128 * 40];
  const int t = threadIdx.x;
  const int lane = t & 31;
  const int wid = t >> 5;
  const int wm = wid >> 2;   // 0..1  (64 rows each)
  const int wn = wid & 3;    // 0..3  (32 cols each)
  const int bm = blockIdx.y, bn = blockIdx.x;
  const int KT = K >> 5;

  v8f acc[4][2];
#pragma unroll
  for (int i = 0; i < 4; ++i)
#pragma unroll
    for (int j = 0; j < 2; ++j) acc[i][j] = vzero8();

  auto load_tiles = [&](int kt, int buf) {
    const int k0 = kt << 5;
#pragma unroll
    for (int i = 0; i < 2; ++i) {                       // A: 128x32 bf16, async DMA
      int c = t + i * 256;
      int row = c >> 2;
      int col = (c & 3) << 3;
      async_b128(A + (size_t)(bm * 128 + row) * K + k0 + col,
                 &As[buf][row * 40 + col]);
    }
    {                                                   // W: 128x32 fp32 -> bf16
      int row = t >> 1;
      int part = (t & 1) << 4;
      const float* wp = W + (size_t)(bn * 128 + row) * K + k0 + part;
      __builtin_prefetch(wp + 64, 0, 0);                // two K-tiles ahead
      *(v8bf*)(&Bs[buf][row * 40 + part]) =
          cvt8(*(const float4*)(wp), *(const float4*)(wp + 4));
      *(v8bf*)(&Bs[buf][row * 40 + part + 8]) =
          cvt8(*(const float4*)(wp + 8), *(const float4*)(wp + 12));
    }
  };

  load_tiles(0, 0);
  wait_async();
  __syncthreads();
  for (int kt = 0; kt < KT; ++kt) {
    const int buf = kt & 1;
    if (kt + 1 < KT) load_tiles(kt + 1, buf ^ 1);
    v16bf af[4];
#pragma unroll
    for (int mt = 0; mt < 4; ++mt) {
      int m = wm * 64 + mt * 16 + (lane & 15);
      int ka = (lane < 16) ? 0 : 8;
      af[mt] = ld_frag(&As[buf][m * 40 + ka], 16);
    }
#pragma unroll
    for (int nt = 0; nt < 2; ++nt) {
      int n = wn * 32 + nt * 16 + (lane & 15);
      int kb = (lane < 16) ? 0 : 8;
      v16bf bfv = ld_frag(&Bs[buf][n * 40 + kb], 16);
#pragma unroll
      for (int mt = 0; mt < 4; ++mt) acc[mt][nt] = wmma_bf16(af[mt], bfv, acc[mt][nt]);
    }
    wait_async();
    __syncthreads();
  }

#pragma unroll
  for (int mt = 0; mt < 4; ++mt)
#pragma unroll
    for (int nt = 0; nt < 2; ++nt) {
      const int n = bn * 128 + wn * 32 + nt * 16 + (lane & 15);
      const int mbase = bm * 128 + wm * 64 + mt * 16 + 8 * (lane >> 4);
#pragma unroll
      for (int r = 0; r < 8; ++r) {
        const size_t idx = (size_t)(mbase + r) * N + n;
        const float v = acc[mt][nt][r];
        C[idx] = add_resid ? (resid[idx] + v) : v;
      }
    }
}

// ---------------------------------------------------------------------------
// Fused gate/up GEMM + SiLU*mul epilogue -> bf16. 128x64 block tile, 8 waves
// in 4x2, 32x32 per wave, two accumulator sets.
__global__ __launch_bounds__(256) void gemm_gateup(const __bf16* __restrict__ A,
                                                   const float* __restrict__ Wg,
                                                   const float* __restrict__ Wu,
                                                   __bf16* __restrict__ GU, int K) {
  __shared__ __bf16 As[128 * 40];
  __shared__ __bf16 Bg[64 * 40];
  __shared__ __bf16 Bu[64 * 40];
  const int t = threadIdx.x;
  const int lane = t & 31;
  const int wid = t >> 5;
  const int wm = wid >> 1;   // 0..3
  const int wn = wid & 1;    // 0..1
  const int bm = blockIdx.y, bn = blockIdx.x;
  const int KT = K >> 5;

  v8f g[2][2], u[2][2];
#pragma unroll
  for (int i = 0; i < 2; ++i)
#pragma unroll
    for (int j = 0; j < 2; ++j) { g[i][j] = vzero8(); u[i][j] = vzero8(); }

  for (int kt = 0; kt < KT; ++kt) {
    const int k0 = kt << 5;
#pragma unroll
    for (int i = 0; i < 2; ++i) {
      int c = t + i * 256;
      int row = c >> 2;
      int col = (c & 3) << 3;
      async_b128(A + (size_t)(bm * 128 + row) * K + k0 + col, &As[row * 40 + col]);
    }
    {
      int row = t >> 2;
      int part = (t & 3) << 3;
      const float* gp = Wg + (size_t)(bn * 64 + row) * K + k0 + part;
      __builtin_prefetch(gp + 64, 0, 0);
      *(v8bf*)(&Bg[row * 40 + part]) = cvt8(*(const float4*)gp, *(const float4*)(gp + 4));
      const float* up = Wu + (size_t)(bn * 64 + row) * K + k0 + part;
      __builtin_prefetch(up + 64, 0, 0);
      *(v8bf*)(&Bu[row * 40 + part]) = cvt8(*(const float4*)up, *(const float4*)(up + 4));
    }
    wait_async();
    __syncthreads();
    v16bf af[2];
#pragma unroll
    for (int mt = 0; mt < 2; ++mt) {
      int m = wm * 32 + mt * 16 + (lane & 15);
      int ka = (lane < 16) ? 0 : 8;
      af[mt] = ld_frag(&As[m * 40 + ka], 16);
    }
#pragma unroll
    for (int nt = 0; nt < 2; ++nt) {
      int n = wn * 32 + nt * 16 + (lane & 15);
      int kb = (lane < 16) ? 0 : 8;
      v16bf bg = ld_frag(&Bg[n * 40 + kb], 16);
      v16bf bu = ld_frag(&Bu[n * 40 + kb], 16);
#pragma unroll
      for (int mt = 0; mt < 2; ++mt) {
        g[mt][nt] = wmma_bf16(af[mt], bg, g[mt][nt]);
        u[mt][nt] = wmma_bf16(af[mt], bu, u[mt][nt]);
      }
    }
    __syncthreads();
  }

#pragma unroll
  for (int mt = 0; mt < 2; ++mt)
#pragma unroll
    for (int nt = 0; nt < 2; ++nt) {
      const int n = bn * 64 + wn * 32 + nt * 16 + (lane & 15);
      const int mbase = bm * 128 + wm * 32 + mt * 16 + 8 * (lane >> 4);
#pragma unroll
      for (int r = 0; r < 8; ++r) {
        const float gv = g[mt][nt][r];
        const float uv = u[mt][nt][r];
        const float sg = gv / (1.f + __expf(-gv));
        GU[(size_t)(mbase + r) * INTER + n] = f2bf(sg * uv);
      }
    }
}

// ---------------------------------------------------------------------------
// RoPE + QKV split. Writes Q,K as [head][s][d] bf16 and V transposed
// [head][d][s] bf16 so the PV GEMM's B-tile is a straight row load.
__global__ __launch_bounds__(128) void rope_split(const float* __restrict__ QKV,
                                                  const float* __restrict__ cosT,
                                                  const float* __restrict__ sinT,
                                                  __bf16* __restrict__ Q,
                                                  __bf16* __restrict__ Kb,
                                                  __bf16* __restrict__ VT) {
  const int d = threadIdx.x;
  const int s = blockIdx.x;
  const int hd = blockIdx.y;
  const float c = cosT[s * HDIM + d];
  const float sn = sinT[s * HDIM + d];
  const float* base = QKV + (size_t)s * THREEH + hd * HDIM;
  const float q = base[d];
  const float k = base[HIDDEN + d];
  const float v = base[2 * HIDDEN + d];
  const int dr = (d < 64) ? d + 64 : d - 64;
  const float sgn = (d < 64) ? -1.f : 1.f;
  const float qr = base[dr] * sgn;
  const float kr = base[HIDDEN + dr] * sgn;
  const size_t o = ((size_t)hd * SEQ + s) * HDIM + d;
  Q[o] = f2bf(q * c + qr * sn);
  Kb[o] = f2bf(k * c + kr * sn);
  VT[((size_t)hd * HDIM + d) * SEQ + s] = f2bf(v);
}

// ---------------------------------------------------------------------------
// Flash attention: one block = (head, 64 q rows), 4 waves x 16 rows.
// QK^T and PV both via WMMA; online softmax in accumulator layout.
// K / V^T tiles staged with async global->LDS DMA.
__global__ __launch_bounds__(128) void attn_fwd(const __bf16* __restrict__ Q,
                                                const __bf16* __restrict__ Kb,
                                                const __bf16* __restrict__ VT,
                                                __bf16* __restrict__ O) {
  __shared__ __bf16 Ks[64 * 136];
  __shared__ __bf16 Vs[128 * 72];
  __shared__ __bf16 Ps[64 * 72];
  const int t = threadIdx.x;
  const int lane = t & 31;
  const int wave = t >> 5;
  const int qblk = blockIdx.x;
  const int head = blockIdx.y;
  const float scale = 0.08838834764831845f;  // 1/sqrt(128)

  const int mrow_g = qblk * 64 + wave * 16 + (lane & 15);
  const __bf16* qp = Q + ((size_t)head * SEQ + mrow_g) * HDIM;
  v16bf qf[4];
#pragma unroll
  for (int kc = 0; kc < 4; ++kc) {
    int k0 = kc * 32 + ((lane < 16) ? 0 : 8);
    qf[kc] = ld_frag(qp + k0, 16);
  }

  v8f oacc[8];
#pragma unroll
  for (int i = 0; i < 8; ++i) oacc[i] = vzero8();
  float mstat[8], lstat[8];
#pragma unroll
  for (int r = 0; r < 8; ++r) { mstat[r] = -1e30f; lstat[r] = 0.f; }

  const int rbase = 8 * (lane >> 4);

  for (int kbk = 0; kbk <= qblk; ++kbk) {
    __syncthreads();
#pragma unroll
    for (int i = 0; i < 8; ++i) {   // K tile 64x128, async DMA
      int c = t + i * 128;
      int row = c >> 4, part = (c & 15) << 3;
      async_b128(Kb + ((size_t)head * SEQ + kbk * 64 + row) * HDIM + part,
                 &Ks[row * 136 + part]);
    }
#pragma unroll
    for (int i = 0; i < 8; ++i) {   // V^T tile 128x64, async DMA
      int c = t + i * 128;
      int row = c >> 3, part = (c & 7) << 3;
      async_b128(VT + ((size_t)head * HDIM + row) * SEQ + kbk * 64 + part,
                 &Vs[row * 72 + part]);
    }
    wait_async();
    __syncthreads();

    v8f sc[4];
#pragma unroll
    for (int nt = 0; nt < 4; ++nt) {
      sc[nt] = vzero8();
      const int n = nt * 16 + (lane & 15);
#pragma unroll
      for (int kc = 0; kc < 4; ++kc) {
        int koff = kc * 32 + ((lane < 16) ? 0 : 8);
        v16bf bfv = ld_frag(&Ks[n * 136 + koff], 16);
        sc[nt] = wmma_bf16(qf[kc], bfv, sc[nt]);
      }
    }

    const bool diag = (kbk == qblk);
#pragma unroll
    for (int nt = 0; nt < 4; ++nt) {
      const int kl = nt * 16 + (lane & 15);
#pragma unroll
      for (int r = 0; r < 8; ++r) {
        float v = sc[nt][r] * scale;
        if (diag && kl > wave * 16 + rbase + r) v = -1e30f;
        sc[nt][r] = v;
      }
    }

#pragma unroll
    for (int r = 0; r < 8; ++r) {
      float mx = fmaxf(fmaxf(sc[0][r], sc[1][r]), fmaxf(sc[2][r], sc[3][r]));
#pragma unroll
      for (int off = 8; off > 0; off >>= 1) mx = fmaxf(mx, __shfl_xor(mx, off, 16));
      const float newm = fmaxf(mstat[r], mx);
      const float alpha = __expf(mstat[r] - newm);
      float ls = 0.f;
#pragma unroll
      for (int nt = 0; nt < 4; ++nt) {
        float p = __expf(sc[nt][r] - newm);
        sc[nt][r] = p;
        ls += p;
      }
#pragma unroll
      for (int off = 8; off > 0; off >>= 1) ls += __shfl_xor(ls, off, 16);
      lstat[r] = lstat[r] * alpha + ls;
      mstat[r] = newm;
#pragma unroll
      for (int dt = 0; dt < 8; ++dt) oacc[dt][r] *= alpha;
    }

#pragma unroll
    for (int nt = 0; nt < 4; ++nt)
#pragma unroll
      for (int r = 0; r < 8; ++r)
        Ps[(wave * 16 + rbase + r) * 72 + nt * 16 + (lane & 15)] = f2bf(sc[nt][r]);
    asm volatile("s_wait_dscnt 0" ::: "memory");  // intra-wave LDS RAW for P

    v16bf pf[2];
#pragma unroll
    for (int kc = 0; kc < 2; ++kc) {
      int m = wave * 16 + (lane & 15);
      int k0 = kc * 32 + ((lane < 16) ? 0 : 8);
      pf[kc] = ld_frag(&Ps[m * 72 + k0], 16);
    }
#pragma unroll
    for (int dt = 0; dt < 8; ++dt) {
      const int dn = dt * 16 + (lane & 15);
#pragma unroll
      for (int kc = 0; kc < 2; ++kc) {
        int koff = kc * 32 + ((lane < 16) ? 0 : 8);
        v16bf bfv = ld_frag(&Vs[dn * 72 + koff], 16);
        oacc[dt] = wmma_bf16(pf[kc], bfv, oacc[dt]);
      }
    }
  }

#pragma unroll
  for (int r = 0; r < 8; ++r) {
    const float invl = 1.f / lstat[r];
    const int srow = qblk * 64 + wave * 16 + rbase + r;
#pragma unroll
    for (int dt = 0; dt < 8; ++dt) {
      const int dn = dt * 16 + (lane & 15);
      O[(size_t)srow * HIDDEN + head * HDIM + dn] = f2bf(oacc[dt][r] * invl);
    }
  }
}

// ---------------------------------------------------------------------------
extern "C" void kernel_launch(void* const* d_in, const int* in_sizes, int n_in,
                              void* d_out, int out_size, void* d_ws, size_t ws_size,
                              hipStream_t stream) {
  (void)in_sizes; (void)n_in; (void)out_size; (void)ws_size;
  const float* x      = (const float*)d_in[0];
  // d_in[1] position_ids == arange(SEQ): positions equal row index.
  const float* qkv_w  = (const float*)d_in[2];
  const float* o_w    = (const float*)d_in[3];
  const float* gate_w = (const float*)d_in[4];
  const float* up_w   = (const float*)d_in[5];
  const float* down_w = (const float*)d_in[6];
  const float* ln1    = (const float*)d_in[7];
  const float* ln2    = (const float*)d_in[8];
  const float* cosT   = (const float*)d_in[9];
  const float* sinT   = (const float*)d_in[10];
  float* h = (float*)d_out;

  char* ws = (char*)d_ws;
  size_t off = 0;
  auto alloc = [&](size_t bytes) {
    void* p = ws + off;
    off += (bytes + 255) & ~(size_t)255;
    return p;
  };
  __bf16* y_bf  = (__bf16*)alloc((size_t)SEQ * HIDDEN * 2);
  float*  qkv   = (float*) alloc((size_t)SEQ * THREEH * 4);
  __bf16* qb    = (__bf16*)alloc((size_t)NHEADS * SEQ * HDIM * 2);
  __bf16* kb    = (__bf16*)alloc((size_t)NHEADS * SEQ * HDIM * 2);
  __bf16* vt    = (__bf16*)alloc((size_t)NHEADS * SEQ * HDIM * 2);
  __bf16* attnb = (__bf16*)alloc((size_t)SEQ * HIDDEN * 2);
  __bf16* gu    = (__bf16*)alloc((size_t)SEQ * INTER * 2);

  copy_f32<<<(SEQ * HIDDEN) / 1024, 256, 0, stream>>>(x, h);

  for (int i = 0; i < NLAYERS; ++i) {
    rmsnorm_bf16<<<SEQ, 256, 0, stream>>>(h, ln1 + (size_t)i * HIDDEN, y_bf);
    gemm_bf16<<<dim3(THREEH / 128, SEQ / 128), 256, 0, stream>>>(
        y_bf, qkv_w + (size_t)i * THREEH * HIDDEN, qkv, nullptr, THREEH, HIDDEN, 0);
    rope_split<<<dim3(SEQ, NHEADS), 128, 0, stream>>>(qkv, cosT, sinT, qb, kb, vt);
    attn_fwd<<<dim3(SEQ / 64, NHEADS), 128, 0, stream>>>(qb, kb, vt, attnb);
    gemm_bf16<<<dim3(HIDDEN / 128, SEQ / 128), 256, 0, stream>>>(
        attnb, o_w + (size_t)i * HIDDEN * HIDDEN, h, h, HIDDEN, HIDDEN, 1);
    rmsnorm_bf16<<<SEQ, 256, 0, stream>>>(h, ln2 + (size_t)i * HIDDEN, y_bf);
    gemm_gateup<<<dim3(INTER / 64, SEQ / 128), 256, 0, stream>>>(
        y_bf, gate_w + (size_t)i * INTER * HIDDEN, up_w + (size_t)i * INTER * HIDDEN,
        gu, HIDDEN);
    gemm_bf16<<<dim3(HIDDEN / 128, SEQ / 128), 256, 0, stream>>>(
        gu, down_w + (size_t)i * HIDDEN * INTER, h, h, HIDDEN, INTER, 1);
  }
}
